// DistributionalQNetwork_712964571807
// MI455X (gfx1250) — matrix operand order repaired and checked by reading, not compiled, not run on hardware
//
#include <hip/hip_runtime.h>
#include <hip/hip_bf16.h>
#include <math.h>

// ---------- types ----------
typedef __attribute__((ext_vector_type(16))) __bf16 bf16x16;
typedef __attribute__((ext_vector_type(8)))  __bf16 bf16x8;
typedef __attribute__((ext_vector_type(8)))  float  f32x8;

#define BATCH   65536
#define NOBS    48
#define NACT    12
#define KPAD0   64
#define H0DIM   1024
#define H1DIM   512
#define H2DIM   256
#define NATOM   101
#define NPAD3   128
#define VMIN   (-10.0f)
#define VMAX   ( 10.0f)

__device__ __forceinline__ unsigned short f2bf(float f) {
    unsigned int u = __float_as_uint(f);
    unsigned int r = u + 0x7FFFu + ((u >> 16) & 1u);   // round-to-nearest-even
    return (unsigned short)(r >> 16);
}

__device__ __forceinline__ bf16x16 cat8(bf16x8 lo, bf16x8 hi) {
    return __builtin_shufflevector(lo, hi, 0,1,2,3,4,5,6,7,8,9,10,11,12,13,14,15);
}

// ---------- CDNA5 async copy: global -> LDS, 16B per lane, ASYNCcnt-tracked ----------
__device__ __forceinline__ void async_cp16(void* lds, const void* g) {
    unsigned l = (unsigned)(uintptr_t)lds;  // flat shared addr[31:0] == LDS byte offset
    asm volatile("global_load_async_to_lds_b128 %0, %1, off"
                 :: "v"(l), "v"(g) : "memory");
}
template<int N>
__device__ __forceinline__ void wait_asynccnt() {
    asm volatile("s_wait_asynccnt %0" :: "i"(N) : "memory");
}

// ---------- packing ----------
__global__ void pack_x_kernel(const float* __restrict__ obs,
                              const float* __restrict__ act,
                              unsigned short* __restrict__ xb) {
    int idx = blockIdx.x * blockDim.x + threadIdx.x;   // B*64
    int row = idx >> 6, col = idx & 63;
    float v = 0.0f;
    if (col < NOBS)             v = obs[row * NOBS + col];
    else if (col < NOBS + NACT) v = act[row * NACT + (col - NOBS)];
    xb[idx] = f2bf(v);
}

// dst = transposed, padded weight: Wt[Np][Kp] from src[K0][N0]
__global__ void pack_wt_kernel(const float* __restrict__ src,
                               unsigned short* __restrict__ dst,
                               int K0, int N0, int Kp, int Np) {
    int idx = blockIdx.x * blockDim.x + threadIdx.x;
    if (idx >= Kp * Np) return;
    int n = idx / Kp, k = idx % Kp;
    float v = (k < K0 && n < N0) ? src[k * N0 + n] : 0.0f;
    dst[idx] = f2bf(v);
}

// ---------- WMMA GEMM:  out[B,N] = relu(A[B,K] * Wt[N,K]^T + bias) ----------
// Block: 256 thr (8 waves). Tile 128x128, K-step 32. Per wave: 32x64 (2x4 wmma).
// Double-buffered LDS, filled by async global->LDS copies (no VGPR round-trip).
#define LDA_S 40   // halfword stride: 80B -> 16B aligned, conflict-free 20-dword bank stride

__device__ __forceinline__ void stage_tiles_async(
    const unsigned short* __restrict__ A, const unsigned short* __restrict__ Wt,
    unsigned short* As, unsigned short* Bs,
    int tid, int rowBase, int colBase, int K, int k0) {
    // both tiles are 128 rows x 32 halfwords = 512 x 16B chunks, 2 each per thread
    #pragma unroll
    for (int i = 0; i < 2; ++i) {
        int c = tid + 256 * i;
        int r = c >> 2, co = (c & 3) * 8;
        async_cp16(&As[r * LDA_S + co], A  + (size_t)(rowBase + r) * K + k0 + co);
        async_cp16(&Bs[r * LDA_S + co], Wt + (size_t)(colBase + r) * K + k0 + co);
    }
}

__device__ __forceinline__ void compute_step(
    const unsigned short* As, const unsigned short* Bs,
    int waveM, int waveN, int m, int h, f32x8 acc[2][4]) {
    // B fragments: K-contiguous per column (lanes 0-15: K0-15, 16-31: K16-31)
    bf16x16 bfr[4];
    #pragma unroll
    for (int ni = 0; ni < 4; ++ni) {
        const unsigned short* p = &Bs[(waveN * 64 + ni * 16 + m) * LDA_S + h * 16];
        bfr[ni] = cat8(*(const bf16x8*)p, *(const bf16x8*)(p + 8));
    }
    // A fragments (lanes 0-15: K0-15, lanes 16-31: K16-31) + 2x4 WMMA
    #pragma unroll
    for (int mi = 0; mi < 2; ++mi) {
        const unsigned short* p = &As[(waveM * 32 + mi * 16 + m) * LDA_S + h * 16];
        bf16x16 afr = cat8(*(const bf16x8*)p, *(const bf16x8*)(p + 8));
        #pragma unroll
        for (int ni = 0; ni < 4; ++ni)
            acc[mi][ni] = __builtin_amdgcn_wmma_f32_16x16x32_bf16(
                false, afr, false, bfr[ni], (short)0, acc[mi][ni], false, false);
    }
}

template<bool RELU, bool OUT_BF16>
__global__ __launch_bounds__(256)
void gemm_wmma_kernel(const unsigned short* __restrict__ A,
                      const unsigned short* __restrict__ Wt,   // [N,K] (transposed)
                      const float* __restrict__ bias, int nBias,
                      void* __restrict__ out, int K, int N) {
    __shared__ unsigned short As[2][128 * LDA_S];
    __shared__ unsigned short Bs[2][128 * LDA_S];

    const int tid  = threadIdx.x;
    const int wid  = tid >> 5, lane = tid & 31;
    const int waveM = wid & 3, waveN = wid >> 2;     // 4x2 waves over 128x128
    const int m = lane & 15, h = lane >> 4;
    const int rowBase = blockIdx.x * 128;
    const int colBase = blockIdx.y * 128;

    f32x8 acc[2][4] = {};

    // prologue: stage tile 0
    stage_tiles_async(A, Wt, As[0], Bs[0], tid, rowBase, colBase, K, 0);

    int buf = 0;
    int k0 = 0;
    // steady state: issue tile t+1, wait for tile t (first 4 of 8 in-flight, in order)
    for (; k0 + 32 < K; k0 += 32, buf ^= 1) {
        stage_tiles_async(A, Wt, As[buf ^ 1], Bs[buf ^ 1], tid, rowBase, colBase, K, k0 + 32);
        if (k0 + 64 < K)   // warm L2 for the tile after next
            __builtin_prefetch(A + (size_t)(rowBase + (tid >> 1)) * K + k0 + 64, 0, 1);
        wait_asynccnt<4>();
        __syncthreads();
        compute_step(As[buf], Bs[buf], waveM, waveN, m, h, acc);
        __syncthreads();
    }
    // final tile
    wait_asynccnt<0>();
    __syncthreads();
    compute_step(As[buf], Bs[buf], waveM, waveN, m, h, acc);

    // --- epilogue: bias + relu + store (C layout: N=lane&15, M=r+8*(lane>>4)) ---
    #pragma unroll
    for (int mi = 0; mi < 2; ++mi)
    #pragma unroll
    for (int ni = 0; ni < 4; ++ni) {
        int col = colBase + waveN * 64 + ni * 16 + (lane & 15);
        float bv = (col < nBias) ? bias[col] : 0.0f;
        #pragma unroll
        for (int r = 0; r < 8; ++r) {
            int row = rowBase + waveM * 32 + mi * 16 + r + 8 * (lane >> 4);
            float v = acc[mi][ni][r] + bv;
            if (RELU) v = fmaxf(v, 0.0f);
            if (OUT_BF16) ((unsigned short*)out)[(size_t)row * N + col] = f2bf(v);
            else          ((float*)out)[(size_t)row * N + col] = v;
        }
    }
}

// ---------- softmax + C51 projection: one wave32 per row ----------
__global__ __launch_bounds__(256)
void proj_kernel(const float* __restrict__ logits,      // [B, NPAD3]
                 const float* __restrict__ rewards,
                 const float* __restrict__ bootstrap,
                 const float* __restrict__ discount,
                 const float* __restrict__ support,     // [NATOM]
                 float* __restrict__ out) {              // [B, NATOM]
    __shared__ float sp[8][104];
    const int wid = threadIdx.x >> 5, lane = threadIdx.x & 31;
    const int row = blockIdx.x * 8 + wid;
    const float* lrow = logits + (size_t)row * NPAD3;

    float x[4];
    float mx = -INFINITY;
    #pragma unroll
    for (int t = 0; t < 4; ++t) {
        int j = lane + 32 * t;
        x[t] = (j < NATOM) ? lrow[j] : -INFINITY;
        mx = fmaxf(mx, x[t]);
    }
    for (int off = 16; off > 0; off >>= 1) mx = fmaxf(mx, __shfl_xor(mx, off, 32));
    float s = 0.0f;
    #pragma unroll
    for (int t = 0; t < 4; ++t) {
        int j = lane + 32 * t;
        x[t] = (j < NATOM) ? __expf(x[t] - mx) : 0.0f;
        s += x[t];
    }
    for (int off = 16; off > 0; off >>= 1) s += __shfl_xor(s, off, 32);
    const float inv = 1.0f / s;

    for (int j = lane; j < NATOM; j += 32) sp[wid][j] = 0.0f;
    __syncthreads();

    const float r  = rewards[row];
    const float bd = bootstrap[row] * discount[row];
    const float dz = (VMAX - VMIN) / (float)(NATOM - 1);
    #pragma unroll
    for (int t = 0; t < 4; ++t) {
        int j = lane + 32 * t;
        if (j < NATOM) {
            float p  = x[t] * inv;
            float tz = fminf(fmaxf(r + bd * support[j], VMIN), VMAX);
            float bp = (tz - VMIN) / dz;
            int l = (int)floorf(bp);
            int u = (int)ceilf(bp);
            bool lm = (u > 0) && (l == u);
            bool um = (l < NATOM - 1) && (l == u);
            int ll = lm ? l - 1 : l;
            int uu = um ? u + 1 : u;
            atomicAdd(&sp[wid][ll], p * ((float)uu - bp));
            atomicAdd(&sp[wid][uu], p * (bp - (float)ll));
        }
    }
    __syncthreads();
    for (int j = lane; j < NATOM; j += 32)
        out[(size_t)row * NATOM + j] = sp[wid][j];
}

// ---------- launch ----------
extern "C" void kernel_launch(void* const* d_in, const int* in_sizes, int n_in,
                              void* d_out, int out_size, void* d_ws, size_t ws_size,
                              hipStream_t stream) {
    const float* obs       = (const float*)d_in[0];
    const float* actions   = (const float*)d_in[1];
    const float* rewards   = (const float*)d_in[2];
    const float* bootstrap = (const float*)d_in[3];
    const float* discount  = (const float*)d_in[4];
    const float* q_support = (const float*)d_in[5];
    const float* W0 = (const float*)d_in[6];  const float* b0 = (const float*)d_in[7];
    const float* W1 = (const float*)d_in[8];  const float* b1 = (const float*)d_in[9];
    const float* W2 = (const float*)d_in[10]; const float* b2 = (const float*)d_in[11];
    const float* W3 = (const float*)d_in[12]; const float* b3 = (const float*)d_in[13];

    char* ws = (char*)d_ws;
    // workspace layout (bytes)
    unsigned short* Xb  = (unsigned short*)(ws + 0);                    // 8 MB
    unsigned short* W0t = (unsigned short*)(ws + (8u  << 20));          // 128 KB
    unsigned short* W1t = (unsigned short*)(ws + (8u  << 20) + 131072 + 0);
    unsigned short* W2t = (unsigned short*)(ws + (8u  << 20) + 131072 + 1048576);
    unsigned short* W3t = (unsigned short*)(ws + (8u  << 20) + 131072 + 1048576 + 262144);
    unsigned short* H0  = (unsigned short*)(ws + (size_t)(16u  << 20)); // 128 MB
    unsigned short* H1  = (unsigned short*)(ws + (size_t)(144u << 20)); // 64 MB
    unsigned short* H2  = (unsigned short*)(ws + (size_t)(208u << 20)); // 32 MB
    float*          LG  = (float*)         (ws + (size_t)(16u  << 20)); // reuse H0 region (H0 dead)

    // pack inputs / transposed weights to bf16 with padding
    pack_x_kernel<<<(BATCH * KPAD0) / 256, 256, 0, stream>>>(obs, actions, Xb);
    pack_wt_kernel<<<(KPAD0 * H0DIM) / 256, 256, 0, stream>>>(W0, W0t, NOBS + NACT, H0DIM, KPAD0, H0DIM);
    pack_wt_kernel<<<(H0DIM * H1DIM) / 256, 256, 0, stream>>>(W1, W1t, H0DIM, H1DIM, H0DIM, H1DIM);
    pack_wt_kernel<<<(H1DIM * H2DIM) / 256, 256, 0, stream>>>(W2, W2t, H1DIM, H2DIM, H1DIM, H2DIM);
    pack_wt_kernel<<<(H2DIM * NPAD3) / 256, 256, 0, stream>>>(W3, W3t, H2DIM, NATOM, H2DIM, NPAD3);

    // MLP via bf16 WMMA GEMMs (f32 accumulate), async-double-buffered staging
    gemm_wmma_kernel<true,  true ><<<dim3(BATCH / 128, H0DIM / 128), 256, 0, stream>>>(
        Xb, W0t, b0, H0DIM, H0, KPAD0, H0DIM);
    gemm_wmma_kernel<true,  true ><<<dim3(BATCH / 128, H1DIM / 128), 256, 0, stream>>>(
        H0, W1t, b1, H1DIM, H1, H0DIM, H1DIM);
    gemm_wmma_kernel<true,  true ><<<dim3(BATCH / 128, H2DIM / 128), 256, 0, stream>>>(
        H1, W2t, b2, H2DIM, H2, H1DIM, H2DIM);
    gemm_wmma_kernel<false, false><<<dim3(BATCH / 128, NPAD3 / 128), 256, 0, stream>>>(
        H2, W3t, b3, NATOM, LG, H2DIM, NPAD3);

    // softmax + C51 projection
    proj_kernel<<<BATCH / 8, 256, 0, stream>>>(LG, rewards, bootstrap, discount, q_support,
                                               (float*)d_out);
}
